// LstmAeUncertaintyModel_39006892982741
// MI455X (gfx1250) — matrix operand order repaired and checked by reading, compiled, not run on hardware
//
#include <hip/hip_runtime.h>

typedef __attribute__((ext_vector_type(16))) __bf16 v16bf;
typedef __attribute__((ext_vector_type(8)))  float  v8f;

union Frag32B {
  uint4 u[2];
  v16bf v;
};

#define BATCH   256
#define HID     512
#define GATES   2048
#define INP     8
#define SEQLEN  512
#define HORIZON 96

__device__ __forceinline__ unsigned short f2bf(float f) {
  union { float f; unsigned u; } v; v.f = f;
  unsigned r = v.u + 0x7FFFu + ((v.u >> 16) & 1u);   // round-to-nearest-even
  return (unsigned short)(r >> 16);
}
__device__ __forceinline__ float bf2f(unsigned short b) {
  union { unsigned u; float f; } v; v.u = ((unsigned)b) << 16;
  return v.f;
}
__device__ __forceinline__ float sigmoidf(float x) { return 1.0f / (1.0f + __expf(-x)); }

__global__ __launch_bounds__(256) void cvt_bf16_kernel(const float* __restrict__ s,
                                                       unsigned short* __restrict__ d, int n) {
  for (int i = blockIdx.x * blockDim.x + threadIdx.x; i < n; i += gridDim.x * blockDim.x)
    d[i] = f2bf(s[i]);
}

__global__ __launch_bounds__(256) void zero_state_kernel(unsigned short* __restrict__ h,
                                                         float* __restrict__ c, int n) {
  for (int i = blockIdx.x * blockDim.x + threadIdx.x; i < n; i += gridDim.x * blockDim.x) {
    h[i] = 0;
    c[i] = 0.0f;
  }
}

// One LSTM time step, fully fused:
//   g = x @ Wih^T + h_in @ Whh^T + bih + bhh           (Whh part via bf16 WMMA)
//   c  = sigmoid(f)*c + sigmoid(i)*tanh(g)
//   h_out = sigmoid(o)*tanh(c)                          (h_out stored bf16)
// Block tile: 32 batch rows x 64 hidden cols (x 4 gates). 8 waves (2x4),
// each wave: one 16x16 M/N position, 4 gate accumulators, K-loop over 512.
__global__ __launch_bounds__(256) void lstm_step_kernel(
    const float* __restrict__ x, int x_row_stride,
    const float* __restrict__ Wih,              // (2048, 8) f32
    const float* __restrict__ bih,
    const float* __restrict__ bhh,
    const unsigned short* __restrict__ Whh,     // (2048, 512) bf16, row-major (K contiguous)
    const unsigned short* __restrict__ h_in,    // (256, 512) bf16
    float* __restrict__ c,                      // (256, 512) f32
    unsigned short* __restrict__ h_out)         // (256, 512) bf16
{
  __shared__ __align__(16) unsigned short a_lds[32 * HID]; // 32 KB h tile
  __shared__ float x_lds[32 * INP];                        // 1 KB x tile

  const int tid = threadIdx.x;
  const int m0  = blockIdx.x * 32;   // batch tile base
  const int n0  = blockIdx.y * 64;   // hidden tile base

  // Stage 32 rows of h_in into LDS: 32*512 bf16 = 2048 uint4, 8 per thread.
  {
    const uint4* src = reinterpret_cast<const uint4*>(h_in + (size_t)m0 * HID);
    uint4*       dst = reinterpret_cast<uint4*>(a_lds);
#pragma unroll
    for (int j = 0; j < 8; ++j)
      dst[tid + j * 256] = src[tid + j * 256];
  }
  // Stage x tile (32 x 8 f32), one element per thread.
  {
    int r = tid / INP, k = tid % INP;
    x_lds[tid] = x[(size_t)(m0 + r) * x_row_stride + k];
  }
  __syncthreads();

  const int wave = tid >> 5;
  const int lane = tid & 31;
  const int half = lane >> 4;        // 0 or 1
  const int lh   = lane & 15;        // 0..15
  const int mw   = (wave >> 2) * 16; // 0 or 16  (row offset within block tile)
  const int nw   = (wave & 3) * 16;  // 0,16,32,48 (hidden-col offset)

  const int col = n0 + nw + lh;      // this lane's hidden column (0..511)

  v8f acc[4];
#pragma unroll
  for (int g = 0; g < 4; ++g) {
    v8f z = {0.f, 0.f, 0.f, 0.f, 0.f, 0.f, 0.f, 0.f};
    acc[g] = z;
  }

  const unsigned short* brow[4];
#pragma unroll
  for (int g = 0; g < 4; ++g)
    brow[g] = Whh + (size_t)(g * HID + col) * HID;

  for (int k0 = 0; k0 < HID; k0 += 32) {
    // A fragment (16x32 bf16): lane<16 holds M=lh, K = k0..k0+7 and k0+16..23;
    // lane>=16 same M, K shifted by 8.
    Frag32B af;
    {
      const uint4* ap = reinterpret_cast<const uint4*>(
          a_lds + (size_t)(mw + lh) * HID + k0 + half * 8);
      af.u[0] = ap[0];
      af.u[1] = ap[2]; // +16 elements = +32 bytes
    }
#pragma unroll
    for (int g = 0; g < 4; ++g) {
      // B fragment (32x16 bf16): lane n holds column N=n, K = k0..k0+7 and k0+16..23;
      // lanes>=16 same N, K shifted by 8 (matches A).
      Frag32B bf_;
      const uint4* bp = reinterpret_cast<const uint4*>(brow[g] + k0 + half * 8);
      bf_.u[0] = bp[0];
      bf_.u[1] = bp[2];
      if (k0 + 32 < HID)
        __builtin_prefetch(brow[g] + k0 + 32 + half * 16, 0, 1);
      acc[g] = __builtin_amdgcn_wmma_f32_16x16x32_bf16(
          false, af.v, false, bf_.v, (short)0, acc[g], false, false);
    }
  }

  // Epilogue: fold in the K=8 input-gate GEMM + biases, then LSTM cell.
  float wih[4][8];
  float bsum[4];
#pragma unroll
  for (int g = 0; g < 4; ++g) {
    int gc = g * HID + col;
    bsum[g] = bih[gc] + bhh[gc];
#pragma unroll
    for (int k = 0; k < 8; ++k)
      wih[g][k] = Wih[gc * INP + k];
  }

#pragma unroll
  for (int r = 0; r < 8; ++r) {
    // C layout: VGPR r, lanes 0-15 -> M=r ; lanes 16-31 -> M=8+r
    int mloc = mw + half * 8 + r;
    int m    = m0 + mloc;
    float gate[4];
#pragma unroll
    for (int g = 0; g < 4; ++g) {
      float s = bsum[g] + acc[g][r];
#pragma unroll
      for (int k = 0; k < 8; ++k)
        s += x_lds[mloc * INP + k] * wih[g][k];
      gate[g] = s;
    }
    float iv = sigmoidf(gate[0]);
    float fv = sigmoidf(gate[1]);
    float gv = tanhf(gate[2]);
    float ov = sigmoidf(gate[3]);
    size_t idx = (size_t)m * HID + col;
    float cn = fv * c[idx] + iv * gv;
    c[idx] = cn;
    h_out[idx] = f2bf(ov * tanhf(cn));
  }
}

// out[b, i] = relu(h[b,:]) . lin_W[i,:] + lin_b[i]   (256x8 outputs, tiny)
__global__ __launch_bounds__(256) void out_linear_kernel(
    const unsigned short* __restrict__ h,   // (256,512) bf16
    const float* __restrict__ W,            // (8,512)
    const float* __restrict__ b,
    float* __restrict__ out, int out_row_stride)
{
  int gid = blockIdx.x * blockDim.x + threadIdx.x;  // 0..2047
  int bi  = gid >> 3;
  int i   = gid & 7;
  const unsigned short* hr = h + (size_t)bi * HID;
  const float*          wr = W + (size_t)i * HID;
  float s = b[i];
  for (int n = 0; n < HID; ++n) {
    float hv = bf2f(hr[n]);
    hv = hv > 0.0f ? hv : 0.0f;
    s += hv * wr[n];
  }
  out[(size_t)bi * out_row_stride + i] = s;
}

extern "C" void kernel_launch(void* const* d_in, const int* in_sizes, int n_in,
                              void* d_out, int out_size, void* d_ws, size_t ws_size,
                              hipStream_t stream) {
  const float* seq     = (const float*)d_in[0];
  const float* enc_Wih = (const float*)d_in[1];
  const float* enc_Whh = (const float*)d_in[2];
  const float* enc_bih = (const float*)d_in[3];
  const float* enc_bhh = (const float*)d_in[4];
  const float* dec_Wih = (const float*)d_in[5];
  const float* dec_Whh = (const float*)d_in[6];
  const float* dec_bih = (const float*)d_in[7];
  const float* dec_bhh = (const float*)d_in[8];
  const float* lin_W   = (const float*)d_in[9];
  const float* lin_b   = (const float*)d_in[10];
  float* out = (float*)d_out;

  // Workspace layout (~5.25 MB): h double-buffer (bf16), c (f32), bf16 weights.
  char* ws = (char*)d_ws;
  unsigned short* h0 = (unsigned short*)ws; ws += (size_t)BATCH * HID * 2;
  unsigned short* h1 = (unsigned short*)ws; ws += (size_t)BATCH * HID * 2;
  float*          cS = (float*)ws;          ws += (size_t)BATCH * HID * 4;
  unsigned short* eW = (unsigned short*)ws; ws += (size_t)GATES * HID * 2;
  unsigned short* dW = (unsigned short*)ws; ws += (size_t)GATES * HID * 2;

  cvt_bf16_kernel<<<512, 256, 0, stream>>>(enc_Whh, eW, GATES * HID);
  cvt_bf16_kernel<<<512, 256, 0, stream>>>(dec_Whh, dW, GATES * HID);
  zero_state_kernel<<<256, 256, 0, stream>>>(h0, cS, BATCH * HID);

  dim3 grid(BATCH / 32, HID / 64);  // 8 x 8 = 64 blocks per step
  unsigned short* hin  = h0;
  unsigned short* hout = h1;

  // Encoder scan: x_t = seq[:, t, :], batch row stride S*I.
  for (int t = 0; t < SEQLEN; ++t) {
    lstm_step_kernel<<<grid, 256, 0, stream>>>(
        seq + (size_t)t * INP, SEQLEN * INP,
        enc_Wih, enc_bih, enc_bhh, eW, hin, cS, hout);
    unsigned short* tmp = hin; hin = hout; hout = tmp;
  }

  // Decoder scan: x_0 = seq[:, S-1, :], then x_t = out[:, t-1, :].
  for (int t = 0; t < HORIZON; ++t) {
    const float* x;
    int xs;
    if (t == 0) { x = seq + (size_t)(SEQLEN - 1) * INP; xs = SEQLEN * INP; }
    else        { x = out + (size_t)(t - 1) * INP;      xs = HORIZON * INP; }
    lstm_step_kernel<<<grid, 256, 0, stream>>>(
        x, xs, dec_Wih, dec_bih, dec_bhh, dW, hin, cS, hout);
    unsigned short* tmp = hin; hin = hout; hout = tmp;
    out_linear_kernel<<<(BATCH * INP) / 256, 256, 0, stream>>>(
        hin, lin_W, lin_b, out + (size_t)t * INP, HORIZON * INP);
  }

  (void)in_sizes; (void)n_in; (void)out_size; (void)ws_size;
}
